// GNNSAGPool_37744172597493
// MI455X (gfx1250) — compile-verified, matrix operand order checked by reading, not run on hardware
//
#include <hip/hip_runtime.h>
#include <hip/hip_bf16.h>

// ---------------- types for WMMA fragments ----------------
typedef __attribute__((ext_vector_type(16))) __bf16 v16bf;
typedef __attribute__((ext_vector_type(8)))  float  v8f;

#define HDIM 256

// ---------------- conversion kernels ----------------
__global__ __launch_bounds__(256) void f32_to_bf16_k(const float* __restrict__ in,
                                                     __bf16* __restrict__ out, int n) {
    int i = blockIdx.x * 256 + threadIdx.x;
    if (i < n) out[i] = (__bf16)in[i];
}

// W (K=256 x N=256, row-major) -> Bt (N x K) bf16, so B-fragment loads are contiguous
__global__ __launch_bounds__(256) void w_to_bf16_t_k(const float* __restrict__ W,
                                                     __bf16* __restrict__ Bt) {
    int k = blockIdx.x;      // 0..255
    int n = threadIdx.x;     // 0..255
    Bt[n * HDIM + k] = (__bf16)W[k * HDIM + n];
}

// ---------------- WMMA GEMM: D(M x 256) = A(M x 256) * B(256 x 256) ----------------
// A row-major bf16; Bt holds B transposed (row n = column n of B), bf16.
// One wave computes a 16x64 output strip (4 N-tiles): the A fragment loaded per
// K-step is reused across 4 v_wmma_f32_16x16x32_bf16 issues, cutting HBM A-traffic 4x.
__global__ __launch_bounds__(32) void wmma_gemm_bf16_k(const __bf16* __restrict__ A,
                                                       const __bf16* __restrict__ Bt,
                                                       float* __restrict__ D, int M) {
    const int tileM  = blockIdx.x;
    const int stripN = blockIdx.y;           // 4 strips of 64 across Nout=256
    const int lane   = threadIdx.x;          // wave32, EXEC all ones
    const int half   = lane >> 4;
    const int l15    = lane & 15;

    const __bf16* arow = A + (size_t)(tileM * 16 + l15) * HDIM;
    const __bf16* brow[4];
#pragma unroll
    for (int j = 0; j < 4; ++j)
        brow[j] = Bt + (size_t)(stripN * 64 + j * 16 + l15) * HDIM;

    v8f acc[4] = {{}, {}, {}, {}};
#pragma unroll
    for (int k0 = 0; k0 < HDIM; k0 += 32) {
        const int kA = k0 + half * 8;        // elements 0..7 : K = kA + e
        const int kB = k0 + 16 + half * 8;   // elements 8..15: K = kB + e
        v16bf a;
#pragma unroll
        for (int e = 0; e < 8; ++e) { a[e] = arow[kA + e]; a[e + 8] = arow[kB + e]; }
#pragma unroll
        for (int j = 0; j < 4; ++j) {
            v16bf b;
#pragma unroll
            for (int e = 0; e < 8; ++e) { b[e] = brow[j][kA + e]; b[e + 8] = brow[j][kB + e]; }
            acc[j] = __builtin_amdgcn_wmma_f32_16x16x32_bf16(false, a, false, b,
                                                             (short)0, acc[j], false, false);
        }
    }
    // C/D layout: VGPR r -> M = half*8 + r, N = lane&15
    const int mbase = tileM * 16 + half * 8;
#pragma unroll
    for (int j = 0; j < 4; ++j) {
        const int ncol = stripN * 64 + j * 16 + l15;
#pragma unroll
        for (int r = 0; r < 8; ++r)
            D[(size_t)(mbase + r) * HDIM + ncol] = acc[j][r];
    }
}

// ---------------- degree / normalization ----------------
__global__ __launch_bounds__(256) void deg_accum_k(const int* __restrict__ dst,
                                                   const float* __restrict__ ew,
                                                   float* __restrict__ deg, int E) {
    int e = blockIdx.x * 256 + threadIdx.x;
    if (e < E) {
        float w = ew ? ew[e] : 1.0f;
        if (w != 0.0f) atomicAdd(&deg[dst[e]], w);
    }
}

__global__ __launch_bounds__(256) void make_dinv_k(const float* __restrict__ deg,
                                                   float* __restrict__ dinv, int n) {
    int i = blockIdx.x * 256 + threadIdx.x;
    if (i < n) dinv[i] = rsqrtf(deg[i] + 2.0f);   // SELF_W = 2
}

// ---------------- edge aggregation (feature dim = 256) ----------------
__global__ __launch_bounds__(256) void edge_agg_feat_k(const float* __restrict__ h,
                                                       const int* __restrict__ src,
                                                       const int* __restrict__ dst,
                                                       const float* __restrict__ ew,
                                                       const float* __restrict__ dinv,
                                                       float* __restrict__ agg, int E) {
    int e = blockIdx.x;
    float w = ew ? ew[e] : 1.0f;
    if (w == 0.0f) return;
    int s = src[e], d = dst[e];
    float norm = dinv[s] * dinv[d] * w;
    int t = threadIdx.x;
    atomicAdd(&agg[(size_t)d * HDIM + t], h[(size_t)s * HDIM + t] * norm);
}

// out = agg + h * (2*dinv^2) + b  (optional ELU), in-place into agg
__global__ __launch_bounds__(256) void self_bias_act_k(float* __restrict__ agg,
                                                       const float* __restrict__ h,
                                                       const float* __restrict__ dinv,
                                                       const float* __restrict__ b,
                                                       int do_elu) {
    size_t i = (size_t)blockIdx.x * 256 + threadIdx.x;
    int node = (int)(i >> 8);
    int f = (int)(i & 255);
    float dv = dinv[node];
    float v = agg[i] + h[i] * (2.0f * dv * dv) + b[f];
    if (do_elu) v = v > 0.0f ? v : expm1f(v);
    agg[i] = v;
}

// ---------------- mat-vec: sv[node] = dot(X[node,:], w) ----------------
__global__ __launch_bounds__(256) void matvec256_k(const float* __restrict__ X,
                                                   const float* __restrict__ w,
                                                   float* __restrict__ out) {
    __shared__ float red[256];
    int node = blockIdx.x, t = threadIdx.x;
    red[t] = X[(size_t)node * HDIM + t] * w[t];
    __syncthreads();
    for (int s = 128; s > 0; s >>= 1) {
        if (t < s) red[t] += red[t + s];
        __syncthreads();
    }
    if (t == 0) out[node] = red[0];
}

__global__ __launch_bounds__(256) void edge_agg_scalar_k(const float* __restrict__ sv,
                                                         const int* __restrict__ src,
                                                         const int* __restrict__ dst,
                                                         const float* __restrict__ ew,
                                                         const float* __restrict__ dinv,
                                                         float* __restrict__ sagg, int E) {
    int e = blockIdx.x * 256 + threadIdx.x;
    if (e < E) {
        float w = ew ? ew[e] : 1.0f;
        if (w != 0.0f) {
            int s = src[e], d = dst[e];
            atomicAdd(&sagg[d], sv[s] * dinv[s] * dinv[d] * w);
        }
    }
}

__global__ __launch_bounds__(256) void score_final_k(const float* __restrict__ sagg,
                                                     const float* __restrict__ sv,
                                                     const float* __restrict__ dinv,
                                                     const float* __restrict__ bp,
                                                     float* __restrict__ score, int n) {
    int i = blockIdx.x * 256 + threadIdx.x;
    if (i < n) {
        float dv = dinv[i];
        score[i] = sagg[i] + sv[i] * (2.0f * dv * dv) + bp[0];
    }
}

// ---------------- exact ranks (sorting-free top-k with jax tie-breaking) ----------------
__global__ __launch_bounds__(256) void rank_k(const float* __restrict__ score, int n,
                                              int* __restrict__ rank) {
    __shared__ float tile[256];
    int i = blockIdx.x * 256 + threadIdx.x;
    float si = (i < n) ? score[i] : 0.0f;
    int cnt = 0;
    for (int base = 0; base < n; base += 256) {
        int j = base + threadIdx.x;
        tile[threadIdx.x] = (j < n) ? score[j] : -3.4e38f;
        __syncthreads();
        int lim = min(256, n - base);
        for (int jj = 0; jj < lim; ++jj) {
            float sj = tile[jj];
            int j2 = base + jj;
            cnt += (sj > si) || (sj == si && j2 < i);
        }
        __syncthreads();
    }
    if (i < n) rank[i] = cnt;
}

__global__ __launch_bounds__(256) void scatter_topk_k(const float* __restrict__ score,
                                                      const int* __restrict__ rank, int n, int k,
                                                      int* __restrict__ perm,
                                                      float* __restrict__ topv) {
    int i = blockIdx.x * 256 + threadIdx.x;
    if (i < n) {
        int r = rank[i];
        if (r < k) { perm[r] = i; topv[r] = score[i]; }
    }
}

__global__ __launch_bounds__(256) void pool_gather_k(const float* __restrict__ xin,
                                                     const int* __restrict__ perm,
                                                     const float* __restrict__ topv,
                                                     float* __restrict__ xout) {
    int r = blockIdx.x, t = threadIdx.x;
    float g = tanhf(topv[r]);
    xout[(size_t)r * HDIM + t] = xin[(size_t)perm[r] * HDIM + t] * g;
}

__global__ __launch_bounds__(256) void edge_remap_k(const int* __restrict__ src,
                                                    const int* __restrict__ dst,
                                                    const float* __restrict__ ew,
                                                    const int* __restrict__ rank, int k,
                                                    int* __restrict__ nsrc, int* __restrict__ ndst,
                                                    float* __restrict__ new_ew, int E) {
    int e = blockIdx.x * 256 + threadIdx.x;
    if (e < E) {
        int rs = rank[src[e]], rd = rank[dst[e]];
        bool ks = rs < k, kd = rd < k;
        nsrc[e] = ks ? rs : 0;
        ndst[e] = kd ? rd : 0;
        float w = ew ? ew[e] : 1.0f;
        new_ew[e] = (ks && kd) ? w : 0.0f;
    }
}

// ---------------- final MLP head: (1x30)->ELU(256)->10 ----------------
__global__ __launch_bounds__(256) void mlp_head_k(const float* __restrict__ topv,
                                                  const float* __restrict__ Wc1,
                                                  const float* __restrict__ bc1,
                                                  const float* __restrict__ Wc2,
                                                  const float* __restrict__ bc2,
                                                  float* __restrict__ out) {
    __shared__ float vals[32];
    __shared__ float hid[256];
    int t = threadIdx.x;
    if (t < 30) vals[t] = topv[t];
    __syncthreads();
    float acc = bc1[t];
    for (int j = 0; j < 30; ++j) acc += vals[j] * Wc1[j * HDIM + t];
    hid[t] = acc > 0.0f ? acc : expm1f(acc);
    __syncthreads();
    if (t < 10) {
        float o = bc2[t];
        for (int i = 0; i < HDIM; ++i) o += hid[i] * Wc2[i * 10 + t];
        out[t] = o;
    }
}

// =====================================================================
extern "C" void kernel_launch(void* const* d_in, const int* in_sizes, int n_in,
                              void* d_out, int out_size, void* d_ws, size_t ws_size,
                              hipStream_t stream) {
    const float* x   = (const float*)d_in[0];
    const int*   ei  = (const int*)d_in[1];
    const float* W1  = (const float*)d_in[2];
    const float* b1  = (const float*)d_in[3];
    const float* Wp1 = (const float*)d_in[4];
    const float* bp1 = (const float*)d_in[5];
    const float* W2  = (const float*)d_in[6];
    const float* b2  = (const float*)d_in[7];
    const float* Wp2 = (const float*)d_in[8];
    const float* bp2 = (const float*)d_in[9];
    const float* W3  = (const float*)d_in[10];
    const float* b3  = (const float*)d_in[11];
    const float* Wc1 = (const float*)d_in[12];
    const float* bc1 = (const float*)d_in[13];
    const float* Wc2 = (const float*)d_in[14];
    const float* bc2 = (const float*)d_in[15];

    const int N = in_sizes[0] / HDIM;
    const int E = in_sizes[1] / 2;
    const int k1 = (N + 1) / 2;
    const int k2 = (k1 + 1) / 2;
    const int* src0 = ei;
    const int* dst0 = ei + E;

    // ---- workspace carve ----
    char* ws = (char*)d_ws;
    size_t off = 0;
    auto carve = [&](size_t bytes) -> void* {
        void* p = ws + off;
        off = (off + bytes + 255) & ~(size_t)255;
        return p;
    };
    float*  B0    = (float*) carve((size_t)N * HDIM * 4);   // h1 / x_pool1 / x2
    float*  B1    = (float*) carve((size_t)N * HDIM * 4);   // x1 / h2 / x_pool2
    __bf16* XB    = (__bf16*)carve((size_t)N * HDIM * 2);
    __bf16* WBT   = (__bf16*)carve((size_t)HDIM * HDIM * 2);
    float*  DEG   = (float*) carve((size_t)N * 4);
    float*  DINV  = (float*) carve((size_t)N * 4);
    float*  SV    = (float*) carve((size_t)N * 4);
    float*  SAGG  = (float*) carve((size_t)N * 4);
    float*  SCORE = (float*) carve((size_t)N * 4);
    int*    RANK  = (int*)   carve((size_t)N * 4);
    int*    PERM  = (int*)   carve((size_t)N * 4);
    float*  TOPV  = (float*) carve((size_t)N * 4);
    int*    srcB  = (int*)   carve((size_t)E * 4);
    int*    dstB  = (int*)   carve((size_t)E * 4);
    float*  ewB   = (float*) carve((size_t)E * 4);
    int*    srcC  = (int*)   carve((size_t)E * 4);
    int*    dstC  = (int*)   carve((size_t)E * 4);
    float*  ewC   = (float*) carve((size_t)E * 4);

    const int EB = (E + 255) / 256;

    // ================= Layer 1: x1 = ELU(GCN(x, W1, b1)), ew = 1 =================
    f32_to_bf16_k<<<(N * HDIM + 255) / 256, 256, 0, stream>>>(x, XB, N * HDIM);
    w_to_bf16_t_k<<<HDIM, 256, 0, stream>>>(W1, WBT);
    wmma_gemm_bf16_k<<<dim3(N / 16, HDIM / 64), 32, 0, stream>>>(XB, WBT, B0, N);  // h1

    hipMemsetAsync(DEG, 0, (size_t)N * 4, stream);
    deg_accum_k<<<EB, 256, 0, stream>>>(dst0, nullptr, DEG, E);
    make_dinv_k<<<(N + 255) / 256, 256, 0, stream>>>(DEG, DINV, N);

    hipMemsetAsync(B1, 0, (size_t)N * HDIM * 4, stream);
    edge_agg_feat_k<<<E, 256, 0, stream>>>(B0, src0, dst0, nullptr, DINV, B1, E);
    self_bias_act_k<<<N, 256, 0, stream>>>(B1, B0, DINV, b1, 1);                   // x1

    // ---- SAGPool 1 ----
    matvec256_k<<<N, 256, 0, stream>>>(B1, Wp1, SV);
    hipMemsetAsync(SAGG, 0, (size_t)N * 4, stream);
    edge_agg_scalar_k<<<EB, 256, 0, stream>>>(SV, src0, dst0, nullptr, DINV, SAGG, E);
    score_final_k<<<(N + 255) / 256, 256, 0, stream>>>(SAGG, SV, DINV, bp1, SCORE, N);
    rank_k<<<(N + 255) / 256, 256, 0, stream>>>(SCORE, N, RANK);
    scatter_topk_k<<<(N + 255) / 256, 256, 0, stream>>>(SCORE, RANK, N, k1, PERM, TOPV);
    pool_gather_k<<<k1, 256, 0, stream>>>(B1, PERM, TOPV, B0);                     // x_pool1
    edge_remap_k<<<EB, 256, 0, stream>>>(src0, dst0, nullptr, RANK, k1, srcB, dstB, ewB, E);

    // ================= Layer 2: x2 = ELU(GCN(x_pool1, W2, b2)) =================
    f32_to_bf16_k<<<(k1 * HDIM + 255) / 256, 256, 0, stream>>>(B0, XB, k1 * HDIM);
    w_to_bf16_t_k<<<HDIM, 256, 0, stream>>>(W2, WBT);
    wmma_gemm_bf16_k<<<dim3(k1 / 16, HDIM / 64), 32, 0, stream>>>(XB, WBT, B1, k1); // h2

    hipMemsetAsync(DEG, 0, (size_t)k1 * 4, stream);
    deg_accum_k<<<EB, 256, 0, stream>>>(dstB, ewB, DEG, E);
    make_dinv_k<<<(k1 + 255) / 256, 256, 0, stream>>>(DEG, DINV, k1);

    hipMemsetAsync(B0, 0, (size_t)k1 * HDIM * 4, stream);
    edge_agg_feat_k<<<E, 256, 0, stream>>>(B1, srcB, dstB, ewB, DINV, B0, E);
    self_bias_act_k<<<k1, 256, 0, stream>>>(B0, B1, DINV, b2, 1);                  // x2

    // ---- SAGPool 2 ----
    matvec256_k<<<k1, 256, 0, stream>>>(B0, Wp2, SV);
    hipMemsetAsync(SAGG, 0, (size_t)k1 * 4, stream);
    edge_agg_scalar_k<<<EB, 256, 0, stream>>>(SV, srcB, dstB, ewB, DINV, SAGG, E);
    score_final_k<<<(k1 + 255) / 256, 256, 0, stream>>>(SAGG, SV, DINV, bp2, SCORE, k1);
    rank_k<<<(k1 + 255) / 256, 256, 0, stream>>>(SCORE, k1, RANK);
    scatter_topk_k<<<(k1 + 255) / 256, 256, 0, stream>>>(SCORE, RANK, k1, k2, PERM, TOPV);
    pool_gather_k<<<k2, 256, 0, stream>>>(B0, PERM, TOPV, B1);                     // x_pool2
    edge_remap_k<<<EB, 256, 0, stream>>>(srcB, dstB, ewB, RANK, k2, srcC, dstC, ewC, E);

    // ================= Layer 3: x3 = GCN(x_pool2, W3, b3) (scalar out) =================
    matvec256_k<<<k2, 256, 0, stream>>>(B1, W3, SV);
    hipMemsetAsync(DEG, 0, (size_t)k2 * 4, stream);
    deg_accum_k<<<EB, 256, 0, stream>>>(dstC, ewC, DEG, E);
    make_dinv_k<<<(k2 + 255) / 256, 256, 0, stream>>>(DEG, DINV, k2);
    hipMemsetAsync(SAGG, 0, (size_t)k2 * 4, stream);
    edge_agg_scalar_k<<<EB, 256, 0, stream>>>(SV, srcC, dstC, ewC, DINV, SAGG, E);
    score_final_k<<<(k2 + 255) / 256, 256, 0, stream>>>(SAGG, SV, DINV, b3, SCORE, k2);

    // ---- global top-30 (sorted desc by construction) + MLP head ----
    rank_k<<<(k2 + 255) / 256, 256, 0, stream>>>(SCORE, k2, RANK);
    scatter_topk_k<<<(k2 + 255) / 256, 256, 0, stream>>>(SCORE, RANK, k2, 30, PERM, TOPV);
    mlp_head_k<<<1, 256, 0, stream>>>(TOPV, Wc1, bc1, Wc2, bc2, (float*)d_out);
}
